// NeuralSDEQModel_45586782879833
// MI455X (gfx1250) — matrix operand order repaired and checked by reading, not compiled, not run on hardware
//
#include <hip/hip_runtime.h>
#include <stdint.h>

typedef _Float16 half16 __attribute__((ext_vector_type(16)));
typedef float    float8 __attribute__((ext_vector_type(8)));

#define N_PATHS 65536
#define N_STEPS 512
#define LOG2E 1.4426950408889634f

#if __has_builtin(__builtin_amdgcn_tanhf)
#define HAVE_NATIVE_TANH 1
__device__ __forceinline__ float tanh_native(float x) {
    return __builtin_amdgcn_tanhf(x);            // v_tanh_f32
}
#else
__device__ __forceinline__ float tanh_native(float x) {
    float e = __builtin_amdgcn_exp2f(2.8853900818f * x);
    return 1.0f - 2.0f * __builtin_amdgcn_rcpf(1.0f + e);
}
#endif

__device__ __forceinline__ float gelu_fast(float x) {
#ifdef HAVE_NATIVE_TANH
    // 0.5*x*(1 + tanh(0.79788456*x + 0.035677408*x^3))
    float u  = x * (0.7978845608f + 0.0356774081f * x * x);
    float hx = 0.5f * x;
    return fmaf(hx, tanh_native(u), hx);
#else
    // x * sigmoid(2u); -log2e folded into the cubic constants
    float a = x * (-2.3022185f - 0.1029433f * x * x);
    float e = __builtin_amdgcn_exp2f(a);
    return x * __builtin_amdgcn_rcpf(1.0f + e);
#endif
}

__device__ __forceinline__ float exp_fast(float x) {
    return __builtin_amdgcn_exp2f(LOG2E * x);
}

union H2U   { _Float16 h[2]; uint32_t u; };
union BFrag { uint4 q[2]; half16 v; };

__global__ __launch_bounds__(256) void neural_sde_q_kernel(
    const float* __restrict__ z1, const float* __restrict__ z2,
    const float* __restrict__ W1, const float* __restrict__ b1,
    const float* __restrict__ W2, const float* __restrict__ b2,
    const float* __restrict__ W3, const float* __restrict__ b3,
    const float* __restrict__ init_log_v,
    float* __restrict__ out)
{
    constexpr float DT      = 1.0f / 252.0f;
    const     float SQRT_DT = 0.0629940788f;       // sqrt(1/252)
    constexpr float KAPPA = 2.72f, THETA = -3.5f, SIGMA_P = 0.85f;
    constexpr float RHO = -0.85f;
    const     float SQ1MR2 = 0.5267826876f;        // sqrt(1 - rho^2)
    constexpr float Rr = 0.0373f, LMAX = 3.0f;
    constexpr float LVMIN = -7.0f, LVMAX = 2.0f;

    // Layer-1 weights packed per neuron: {W1[k][0], W1[k][1], b1[k], 0}
    __shared__ __align__(16) float4 w1pack[32];
    __shared__ __align__(16) uint32_t h1buf[8][32][16];  // per-wave 32 paths x 32 f16 feats

    const int tid  = threadIdx.x;
    const int wave = tid >> 5;
    const int lane = tid & 31;

    if (tid < 32) w1pack[tid] = make_float4(W1[2 * tid], W1[2 * tid + 1], b1[tid], 0.0f);
    __syncthreads();

    const int  ncol = lane & 15;
    const bool low  = lane < 16;
    const int  roff = low ? 0 : 8;                 // D-row offset for this half-wave

    // ---- Transposed layer 2: A = W2 (rows = output feature n), loop-invariant ----
    // 16-bit A 16x32 layout: lane m<16 holds K {0..7,16..23}; lane m+16 holds K {8..15,24..31}
    const int k0 = low ? 0 : 8;
    half16 af0, af1;
    #pragma unroll
    for (int j = 0; j < 8; ++j) {
        af0[j]     = (_Float16)W2[ncol * 32 + k0 + j];
        af0[j + 8] = (_Float16)W2[ncol * 32 + k0 + 16 + j];
        af1[j]     = (_Float16)W2[(ncol + 16) * 32 + k0 + j];
        af1[j + 8] = (_Float16)W2[(ncol + 16) * 32 + k0 + 16 + j];
    }
    // Per-lane W3 slices and b2 folded into C accumulators (indexed by feature row)
    float w3a[8], w3b[8];
    float8 cinia, cinib;
    #pragma unroll
    for (int r = 0; r < 8; ++r) {
        w3a[r]   = W3[roff + r];
        w3b[r]   = W3[16 + roff + r];
        cinia[r] = b2[roff + r];
        cinib[r] = b2[16 + roff + r];
    }
    const float b3s = b3[0];

    const long gpath = (long)blockIdx.x * 256 + tid;
    const float* z1p = z1 + gpath * N_STEPS;
    const float* z2p = z2 + gpath * N_STEPS;
    float* lvout = out + gpath * N_STEPS;
    float* spout = out + (long)N_PATHS * N_STEPS + gpath * N_STEPS;

    float log_v  = init_log_v[0];
    float log_s  = 0.0f;
    float lam_sq = 0.0f;

    for (int s0 = 0; s0 < N_STEPS; s0 += 4) {
        const float4 zv = *(const float4*)(z1p + s0);
        const float4 zs = *(const float4*)(z2p + s0);
        const float zva[4] = { zv.x, zv.y, zv.z, zv.w };
        const float zsa[4] = { zs.x, zs.y, zs.z, zs.w };
        float lvst[4], spst[4];

        #pragma unroll
        for (int j = 0; j < 4; ++j) {
            const int   s = s0 + j;
            const float t = (float)s * (1.0f / (float)N_STEPS);

            // ---- layer 1 (K=2) in VALU, pack f16, stage own path row to LDS ----
            uint32_t hpk[16];
            #pragma unroll
            for (int k = 0; k < 16; ++k) {
                const float4 wa = w1pack[2 * k];
                const float4 wb = w1pack[2 * k + 1];
                float p0 = fmaf(wa.x, log_v, fmaf(wa.y, t, wa.z));
                float p1 = fmaf(wb.x, log_v, fmaf(wb.y, t, wb.z));
                H2U u;
                u.h[0] = (_Float16)gelu_fast(p0);
                u.h[1] = (_Float16)gelu_fast(p1);
                hpk[k] = u.u;
            }
            uint4* dst = reinterpret_cast<uint4*>(&h1buf[wave][lane][0]);
            dst[0] = make_uint4(hpk[0],  hpk[1],  hpk[2],  hpk[3]);
            dst[1] = make_uint4(hpk[4],  hpk[5],  hpk[6],  hpk[7]);
            dst[2] = make_uint4(hpk[8],  hpk[9],  hpk[10], hpk[11]);
            dst[3] = make_uint4(hpk[12], hpk[13], hpk[14], hpk[15]);

            // ---- B fragments: lane = path column, K contiguous within path row ----
            // lanes 0-15 carry K={0..7,16..23}, lanes 16-31 carry K={8..15,24..31}
            const int kb = low ? 0 : 1;
            const uint4* rp0 = reinterpret_cast<const uint4*>(&h1buf[wave][ncol][0]);
            const uint4* rp1 = reinterpret_cast<const uint4*>(&h1buf[wave][ncol + 16][0]);
            BFrag bq0, bq1;
            bq0.q[0] = rp0[kb];  bq0.q[1] = rp0[kb + 2];
            bq1.q[0] = rp1[kb];  bq1.q[1] = rp1[kb + 2];

            // ---- layer 2 (transposed): D[feature][path], bias in C ----
            float8 c00 = __builtin_amdgcn_wmma_f32_16x16x32_f16(false, af0, false, bq0.v, (short)0, cinia, false, false);
            float8 c10 = __builtin_amdgcn_wmma_f32_16x16x32_f16(false, af1, false, bq0.v, (short)0, cinib, false, false);
            float8 c01 = __builtin_amdgcn_wmma_f32_16x16x32_f16(false, af0, false, bq1.v, (short)0, cinia, false, false);
            float8 c11 = __builtin_amdgcn_wmma_f32_16x16x32_f16(false, af1, false, bq1.v, (short)0, cinib, false, false);

            // ---- layer 3: per-lane dot over resident features, 1 shuffle to finish ----
            float part0 = 0.0f, part1 = 0.0f;
            #pragma unroll
            for (int r = 0; r < 8; ++r) {
                part0 += w3a[r] * gelu_fast(c00[r]) + w3b[r] * gelu_fast(c10[r]);
                part1 += w3a[r] * gelu_fast(c01[r]) + w3b[r] * gelu_fast(c11[r]);
            }
            part0 += __shfl_xor(part0, 16, 32);
            part1 += __shfl_xor(part1, 16, 32);
            const float raw = (low ? part0 : part1) + b3s;   // this lane's own path
            const float lam = LMAX * tanh_native(raw);

            // ---- Euler step (per-lane registers) ----
            const float dwv = SQRT_DT * zva[j];
            const float dws = RHO * dwv + SQ1MR2 * SQRT_DT * zsa[j];
            const float mu_p    = KAPPA * (THETA - log_v) * DT;
            const float drift_q = mu_p - lam * SIGMA_P * DT;
            float lvn = log_v + drift_q + SIGMA_P * dwv;
            lvn = fminf(fmaxf(lvn, LVMIN), LVMAX);
            const float ev  = exp_fast(log_v);
            const float vol = __builtin_amdgcn_sqrtf(fmaxf(ev, 1e-10f));
            const float lsn = log_s + (Rr - 0.5f * ev) * DT + vol * dws;
            lam_sq += lam * lam * DT;
            log_v = lvn;
            log_s = lsn;
            lvst[j] = lvn;
            spst[j] = exp_fast(lsn);
        }
        *(float4*)(lvout + s0) = make_float4(lvst[0], lvst[1], lvst[2], lvst[3]);
        *(float4*)(spout + s0) = make_float4(spst[0], spst[1], spst[2], spst[3]);
    }
    out[2L * N_PATHS * N_STEPS + gpath] = lam_sq;
}

extern "C" void kernel_launch(void* const* d_in, const int* in_sizes, int n_in,
                              void* d_out, int out_size, void* d_ws, size_t ws_size,
                              hipStream_t stream) {
    const float* z1  = (const float*)d_in[0];
    const float* z2  = (const float*)d_in[1];
    const float* W1  = (const float*)d_in[2];
    const float* b1  = (const float*)d_in[3];
    const float* W2  = (const float*)d_in[4];
    const float* b2  = (const float*)d_in[5];
    const float* W3  = (const float*)d_in[6];
    const float* b3  = (const float*)d_in[7];
    const float* ilv = (const float*)d_in[8];
    float* out = (float*)d_out;

    dim3 grid(N_PATHS / 256);
    dim3 block(256);
    neural_sde_q_kernel<<<grid, block, 0, stream>>>(z1, z2, W1, b1, W2, b2, W3, b3, ilv, out);
}
